// EMAVectorQuantizer_26551487824056
// MI455X (gfx1250) — matrix operand order repaired and checked by reading, not compiled, not run on hardware
//
#include <hip/hip_runtime.h>
#include <hip/hip_bf16.h>

typedef __attribute__((ext_vector_type(16))) __bf16 v16bf;
typedef __attribute__((ext_vector_type(8)))  float  v8f;

#define K_CODES 8192
#define DIM     256
#define N_ROWS  16384
#define DECAYF  0.99f
#define EPSF    1e-5f

// ---------------- prep: affine codebook, f32 + bf16 hi/lo split + row norms ----
__global__ void __launch_bounds__(256) prep_emb_kernel(
    const float* __restrict__ embedding, const float* __restrict__ mean,
    const float* __restrict__ stdv, float* __restrict__ emb_f32,
    __bf16* __restrict__ ehi, __bf16* __restrict__ elo, float* __restrict__ enorm)
{
    const int k = blockIdx.x, d = threadIdx.x;
    const size_t i = (size_t)k * DIM + d;
    const float e = mean[d] + stdv[d] * embedding[i];
    emb_f32[i] = e;
    const __bf16 h = (__bf16)e;
    const float hf = (float)h;
    ehi[i] = h;
    elo[i] = (__bf16)(e - hf);
    __shared__ float s[256];
    s[d] = e * e;
    __syncthreads();
    for (int str = 128; str > 0; str >>= 1) {
        if (d < str) s[d] += s[d + str];
        __syncthreads();
    }
    if (d == 0) enorm[k] = s[0];
}

// ---------------- prep: bf16 hi/lo split of x -------------------------------
__global__ void __launch_bounds__(256) prep_x_kernel(
    const float* __restrict__ x, __bf16* __restrict__ xhi, __bf16* __restrict__ xlo)
{
    const size_t i = (size_t)blockIdx.x * 256 + threadIdx.x;
    const float v = x[i];
    const __bf16 h = (__bf16)v;
    const float hf = (float)h;
    xhi[i] = h;
    xlo[i] = (__bf16)(v - hf);
}

// ---------------- fused distance-GEMM + argmin (WMMA bf16, split precision) --
__global__ void __launch_bounds__(256) vq_argmin_kernel(
    const __bf16* __restrict__ xhi, const __bf16* __restrict__ xlo,
    const __bf16* __restrict__ ehi, const __bf16* __restrict__ elo,
    const float* __restrict__ enorm, int* __restrict__ idx_out,
    float* __restrict__ idx_f_out)
{
    const int lane = threadIdx.x & 31;
    const int wave = threadIdx.x >> 5;
    const int r    = lane & 15;
    const int half = lane >> 4;
    const int m_base = blockIdx.x * 128 + wave * 16;   // 8 waves * 16 rows

    union Frag { v16bf v; uint4 u[2]; };

    // Resident A fragments: row (m_base + r), 8 K-steps of 32, hi and lo parts.
    // A 16x32 bf16 layout: lanes 0-15 hold K {kb..kb+7, kb+16..kb+23},
    //                      lanes 16-31 hold K {kb+8..kb+15, kb+24..kb+31}.
    Frag ah[8], al[8];
    const __bf16* xh_row = xhi + (size_t)(m_base + r) * DIM;
    const __bf16* xl_row = xlo + (size_t)(m_base + r) * DIM;
#pragma unroll
    for (int s = 0; s < 8; ++s) {
        const int kb = s * 32;
        ah[s].u[0] = *(const uint4*)(xh_row + kb + half * 8);
        ah[s].u[1] = *(const uint4*)(xh_row + kb + 16 + half * 8);
        al[s].u[0] = *(const uint4*)(xl_row + kb + half * 8);
        al[s].u[1] = *(const uint4*)(xl_row + kb + 16 + half * 8);
    }

    float best[8];
    int   bidx[8];
#pragma unroll
    for (int j = 0; j < 8; ++j) { best[j] = 3.4e38f; bidx[j] = 0; }

    for (int nt = 0; nt < K_CODES / 16; ++nt) {
        const int n_base = nt * 16;
        // B 32x16 bf16 layout: lane's column = r; lanes 0-15 hold K {kb..kb+7, kb+16..kb+23},
        // lanes 16-31 hold K {kb+8..kb+15, kb+24..kb+31} (matches A).
        const __bf16* eh_row = ehi + (size_t)(n_base + r) * DIM;
        const __bf16* el_row = elo + (size_t)(n_base + r) * DIM;
        v8f acc = {0.f, 0.f, 0.f, 0.f, 0.f, 0.f, 0.f, 0.f};
#pragma unroll
        for (int s = 0; s < 8; ++s) {
            const int kb = s * 32 + half * 8;
            Frag bh, bl;
            bh.u[0] = *(const uint4*)(eh_row + kb);
            bh.u[1] = *(const uint4*)(eh_row + kb + 16);
            bl.u[0] = *(const uint4*)(el_row + kb);
            bl.u[1] = *(const uint4*)(el_row + kb + 16);
            // x.e ~= xh.eh + xh.el + xl.eh   (fp32-class accuracy)
            acc = __builtin_amdgcn_wmma_f32_16x16x32_bf16(
                false, ah[s].v, false, bh.v, (short)0, acc, false, false);
            acc = __builtin_amdgcn_wmma_f32_16x16x32_bf16(
                false, ah[s].v, false, bl.v, (short)0, acc, false, false);
            acc = __builtin_amdgcn_wmma_f32_16x16x32_bf16(
                false, al[s].v, false, bh.v, (short)0, acc, false, false);
        }
        // C layout: lane's column N = r; VGPR j -> row M = j + 8*half.
        // d2 = ||x||^2 - 2 x.e + ||e||^2 ; row-constant term dropped for argmin.
        const float en  = enorm[n_base + r];
        const int ncol  = n_base + r;
#pragma unroll
        for (int j = 0; j < 8; ++j) {
            const float cand = en - 2.0f * acc[j];
            if (cand < best[j]) { best[j] = cand; bidx[j] = ncol; }
        }
    }

    // min-reduce across the 16 lanes holding the same row (same half)
#pragma unroll
    for (int off = 1; off < 16; off <<= 1) {
#pragma unroll
        for (int j = 0; j < 8; ++j) {
            const float ov = __shfl_xor(best[j], off, 32);
            const int   oi = __shfl_xor(bidx[j], off, 32);
            if (ov < best[j] || (ov == best[j] && oi < bidx[j])) {
                best[j] = ov; bidx[j] = oi;
            }
        }
    }

    if (r == 0) {
#pragma unroll
        for (int j = 0; j < 8; ++j) {
            const int m = m_base + half * 8 + j;
            idx_out[m]   = bidx[j];
            idx_f_out[m] = (float)bidx[j];
        }
    }
}

// ---------------- gather quantized, loss partials, scatter dw/counts ---------
__global__ void __launch_bounds__(256) quant_loss_kernel(
    const float* __restrict__ x, const float* __restrict__ emb_f32,
    const int* __restrict__ idx, float* __restrict__ q_out,
    float* __restrict__ loss_partials, float* __restrict__ dw,
    float* __restrict__ counts)
{
    const int nrow = blockIdx.x;
    const int d = threadIdx.x;
    const int k = idx[nrow];
    const size_t xi = (size_t)nrow * DIM + d;
    const float xv = x[xi];
    const float qv = emb_f32[(size_t)k * DIM + d];
    q_out[xi] = qv;
    atomicAdd(&dw[(size_t)k * DIM + d], xv);
    if (d == 0) atomicAdd(&counts[k], 1.0f);
    const float diff = qv - xv;
    __shared__ float s[256];
    s[d] = diff * diff;
    __syncthreads();
    for (int str = 128; str > 0; str >>= 1) {
        if (d < str) s[d] += s[d + str];
        __syncthreads();
    }
    if (d == 0) loss_partials[nrow] = s[0];
}

__global__ void __launch_bounds__(256) loss_final_kernel(
    const float* __restrict__ partials, float* __restrict__ loss_out)
{
    __shared__ float s[256];
    const int t = threadIdx.x;
    float sum = 0.f;
    for (int i = t; i < N_ROWS; i += 256) sum += partials[i];
    s[t] = sum;
    __syncthreads();
    for (int str = 128; str > 0; str >>= 1) {
        if (t < str) s[t] += s[t + str];
        __syncthreads();
    }
    if (t == 0) loss_out[0] = 2.0f * s[0] / (float)(N_ROWS * DIM);
}

// ---------------- EMA cluster-size + partial sums ----------------------------
__global__ void __launch_bounds__(256) ncs_kernel(
    const float* __restrict__ cluster_size, const float* __restrict__ counts,
    float* __restrict__ ncs_out, float* __restrict__ ncs_partials)
{
    const int k = blockIdx.x * 256 + threadIdx.x;
    const float v = cluster_size[k] * DECAYF + (1.0f - DECAYF) * counts[k];
    ncs_out[k] = v;
    __shared__ float s[256];
    s[threadIdx.x] = v;
    __syncthreads();
    for (int str = 128; str > 0; str >>= 1) {
        if (threadIdx.x < str) s[threadIdx.x] += s[threadIdx.x + str];
        __syncthreads();
    }
    if (threadIdx.x == 0) ncs_partials[blockIdx.x] = s[0];
}

__global__ void nsum_kernel(const float* __restrict__ partials, float* __restrict__ nsum)
{
    if (threadIdx.x == 0) {
        float s = 0.f;
        for (int i = 0; i < 32; ++i) s += partials[i];
        nsum[0] = s;
    }
}

// ---------------- EMA weights + smoothed new embedding -----------------------
__global__ void __launch_bounds__(256) ema_emb_kernel(
    const float* __restrict__ ema_w, const float* __restrict__ dw,
    const float* __restrict__ ncs, const float* __restrict__ nsum,
    float* __restrict__ ema_out, float* __restrict__ emb_out)
{
    const int k = blockIdx.x, d = threadIdx.x;
    const size_t i = (size_t)k * DIM + d;
    const float ne = ema_w[i] * DECAYF + (1.0f - DECAYF) * dw[i];
    ema_out[i] = ne;
    const float n = nsum[0];
    const float sm = (ncs[k] + EPSF) / (n + (float)K_CODES * EPSF) * n;
    emb_out[i] = ne / sm;
}

extern "C" void kernel_launch(void* const* d_in, const int* in_sizes, int n_in,
                              void* d_out, int out_size, void* d_ws, size_t ws_size,
                              hipStream_t stream) {
    (void)in_sizes; (void)n_in; (void)out_size; (void)ws_size;
    const float* x           = (const float*)d_in[0];
    const float* embedding   = (const float*)d_in[1];
    const float* affine_mean = (const float*)d_in[2];
    const float* affine_std  = (const float*)d_in[3];
    const float* cluster_sz  = (const float*)d_in[4];
    const float* ema_w       = (const float*)d_in[5];

    // ---- output layout (flat f32, return order) ----
    float* out = (float*)d_out;
    float* out_q    = out;                        // 4194304
    float* out_loss = out + 4194304;              // 1
    float* out_idx  = out + 4194305;              // 16384
    float* out_ncs  = out + 4210689;              // 8192
    float* out_ema  = out + 4218881;              // 2097152
    float* out_emb  = out + 6316033;              // 2097152

    // ---- workspace layout (byte offsets) ----
    char* ws = (char*)d_ws;
    __bf16* x_hi    = (__bf16*)(ws + 0);          //  8 MB
    __bf16* x_lo    = (__bf16*)(ws + 8388608);    //  8 MB
    float*  emb_f32 = (float*)(ws + 16777216);    //  8 MB
    __bf16* e_hi    = (__bf16*)(ws + 25165824);   //  4 MB
    __bf16* e_lo    = (__bf16*)(ws + 29360128);   //  4 MB
    float*  e_norm  = (float*)(ws + 33554432);    // 32 KB
    int*    idx_i32 = (int*)  (ws + 33587200);    // 64 KB
    float*  counts  = (float*)(ws + 33652736);    // 32 KB
    float*  dw      = (float*)(ws + 33685504);    //  8 MB
    float*  loss_p  = (float*)(ws + 42074112);    // 64 KB
    float*  ncs_p   = (float*)(ws + 42139648);    // 128 B
    float*  n_sum   = (float*)(ws + 42139776);    // 4 B

    // zero the scatter targets (graph-capturable)
    hipMemsetAsync(counts, 0, (size_t)K_CODES * 4, stream);
    hipMemsetAsync(dw, 0, (size_t)K_CODES * DIM * 4, stream);

    prep_emb_kernel<<<K_CODES, 256, 0, stream>>>(embedding, affine_mean, affine_std,
                                                 emb_f32, e_hi, e_lo, e_norm);
    prep_x_kernel<<<(N_ROWS * DIM) / 256, 256, 0, stream>>>(x, x_hi, x_lo);

    vq_argmin_kernel<<<N_ROWS / 128, 256, 0, stream>>>(x_hi, x_lo, e_hi, e_lo,
                                                       e_norm, idx_i32, out_idx);

    quant_loss_kernel<<<N_ROWS, 256, 0, stream>>>(x, emb_f32, idx_i32, out_q,
                                                  loss_p, dw, counts);
    loss_final_kernel<<<1, 256, 0, stream>>>(loss_p, out_loss);

    ncs_kernel<<<K_CODES / 256, 256, 0, stream>>>(cluster_sz, counts, out_ncs, ncs_p);
    nsum_kernel<<<1, 32, 0, stream>>>(ncs_p, n_sum);

    ema_emb_kernel<<<K_CODES, 256, 0, stream>>>(ema_w, dw, out_ncs, n_sum,
                                                out_ema, out_emb);
}
